// NLBlockND_68891275427978
// MI455X (gfx1250) — matrix operand order repaired and checked by reading, not compiled, not run on hardware
//
#include <hip/hip_runtime.h>
#include <math.h>

// ---------------- problem constants ----------------
#define BB   8
#define CC   1024
#define CII  512
#define NN   3136          // H*W = 56*56
#define NT   196           // NN/16
#define NG   49            // NN/64
#define BN_EPS 1e-5f

typedef __attribute__((ext_vector_type(16))) __bf16 v16bf;
typedef __attribute__((ext_vector_type(8)))  __bf16 v8bf;
typedef __attribute__((ext_vector_type(8)))  float  v8f;
typedef __attribute__((ext_vector_type(4)))  unsigned int u32x4;
typedef __attribute__((ext_vector_type(8)))  int i32x8;
typedef __attribute__((ext_vector_type(4)))  int i32x4;

#if defined(__has_builtin)
#if __has_builtin(__builtin_amdgcn_tensor_load_to_lds) && __has_builtin(__builtin_amdgcn_s_wait_tensorcnt)
#define HAVE_TDM 1
#endif
#endif

__device__ __forceinline__ __bf16 f2bf(float f) {
  unsigned u = __builtin_bit_cast(unsigned, f);
  unsigned r = u + 0x7fffu + ((u >> 16) & 1u);      // round-to-nearest-even
  return __builtin_bit_cast(__bf16, (unsigned short)(r >> 16));
}

// A-fragment (16x32 bf16) from a row-major [rows x ld] buffer.
// ISA layout: lane L -> row = L&15; elems 0..7 = K k0+8*hi.. ; elems 8..15 = K k0+16+8*hi..
__device__ __forceinline__ v16bf frag_a(const __bf16* __restrict__ base, int ld,
                                        int row, int k0, int hi) {
  const __bf16* p = base + (size_t)row * ld + k0 + hi * 8;
  v8bf lo = *(const v8bf*)(p);
  v8bf hh = *(const v8bf*)(p + 16);
  return __builtin_shufflevector(lo, hh, 0,1,2,3,4,5,6,7,8,9,10,11,12,13,14,15);
}

// B-fragment (32x16 bf16) from a "fragment-major" row-major [cols x ld] buffer.
// ISA layout: lane L -> col = L&15; elems 0..7 = K k0+8*hi.. ; elems 8..15 = K k0+16+8*hi..
__device__ __forceinline__ v16bf frag_b(const __bf16* __restrict__ base, int ld,
                                        int col, int k0, int hi) {
  const __bf16* p = base + (size_t)col * ld + k0 + hi * 8;
  v8bf lo = *(const v8bf*)(p);
  v8bf hh = *(const v8bf*)(p + 16);
  return __builtin_shufflevector(lo, hh, 0,1,2,3,4,5,6,7,8,9,10,11,12,13,14,15);
}

#define WMMA_BF16(a, b, c) \
  __builtin_amdgcn_wmma_f32_16x16x32_bf16(false, (a), false, (b), (short)0, (c), false, false)

// ---------------- K0: pack weights to bf16, build fused QKV weight + bias ----------------
__global__ __launch_bounds__(256) void k0_prep(
    const float* __restrict__ theta_w, const float* __restrict__ phi_w,
    const float* __restrict__ g_w,     const float* __restrict__ theta_b,
    const float* __restrict__ phi_b,   const float* __restrict__ g_b,
    const float* __restrict__ wz_w,
    __bf16* __restrict__ wqkv, __bf16* __restrict__ wzb, float* __restrict__ qkv_b) {
  const int SZ1 = 3 * CII * CC;                      // 1536*1024
  int i = blockIdx.x * blockDim.x + threadIdx.x;
  if (i < SZ1) {
    int o = i >> 10, c = i & (CC - 1);
    float v = (o < CII)     ? theta_w[(size_t)o * CC + c]
            : (o < 2 * CII) ? phi_w[(size_t)(o - CII) * CC + c]
                            : g_w[(size_t)(o - 2 * CII) * CC + c];
    wqkv[i] = f2bf(v);
    if (c == 0)
      qkv_b[o] = (o < CII) ? theta_b[o] : (o < 2 * CII) ? phi_b[o - CII] : g_b[o - 2 * CII];
  } else {
    int j = i - SZ1;                                  // < 1024*512
    wzb[j] = f2bf(wz_w[j]);
  }
}

// ---------------- K1: tiled transpose-cast  x[B,C,N] f32 -> xT[B,N,C] bf16 ----------------
__global__ __launch_bounds__(256) void k1_transpose_cast(
    const float* __restrict__ x, __bf16* __restrict__ xT) {
  __shared__ float tile[32][33];
  const int b  = blockIdx.z;
  const int c0 = blockIdx.x * 32;
  const int n0 = blockIdx.y * 32;
  const int j  = threadIdx.x & 31;
  const int i0 = threadIdx.x >> 5;                    // 0..7
  const float* xb = x + (size_t)b * CC * NN;
#pragma unroll
  for (int ii = 0; ii < 4; ++ii) {
    int i = i0 + ii * 8;
    tile[i][j] = xb[(size_t)(c0 + i) * NN + n0 + j];  // coalesced in n
  }
  __syncthreads();
  __bf16* xtb = xT + (size_t)b * NN * CC;
#pragma unroll
  for (int ii = 0; ii < 4; ++ii) {
    int i = i0 + ii * 8;                              // n-local
    xtb[(size_t)(n0 + i) * CC + c0 + j] = f2bf(tile[j][i]);  // coalesced in c
  }
}

// ---------------- K2: fused QKV projection GEMM ----------------
// Each wave computes a 32(M) x 64(N) macro-tile: 8 accumulators, 2 A-frag + 4 B-frag
// loads per k-step feeding 8 WMMAs (0.75 fragment loads per WMMA).
__global__ __launch_bounds__(256) void k2_qkv_gemm(
    const __bf16* __restrict__ wqkv, const float* __restrict__ qkv_b,
    const __bf16* __restrict__ xT,
    __bf16* __restrict__ thetaT, __bf16* __restrict__ phiT, __bf16* __restrict__ g_cn) {
  const int b    = blockIdx.y;
  const int w    = threadIdx.x >> 5;
  const int lane = threadIdx.x & 31;
  const int hi   = (lane >> 4) & 1;
  const int l15  = lane & 15;
  const int t  = blockIdx.x * 8 + w;                  // [0, 48*49)
  const int ng = t % NG, mg = t / NG;                 // 64-col group, 32-row group
  const __bf16* xTb = xT + (size_t)b * NN * CC;
  const int ar0 = mg * 32 + l15;
  const int nb  = ng * 64;
  v8f acc[2][4];
#pragma unroll
  for (int i = 0; i < 2; ++i)
#pragma unroll
    for (int j = 0; j < 4; ++j) acc[i][j] = (v8f){0.f,0.f,0.f,0.f,0.f,0.f,0.f,0.f};
  for (int k0 = 0; k0 < CC; k0 += 32) {
    __builtin_prefetch(xTb + (size_t)(nb + l15) * CC + k0 + 64, 0, 3);
    v16bf a0 = frag_a(wqkv, CC, ar0,      k0, hi);
    v16bf a1 = frag_a(wqkv, CC, ar0 + 16, k0, hi);
#pragma unroll
    for (int j = 0; j < 4; ++j) {
      v16bf bb = frag_b(xTb, CC, nb + j * 16 + l15, k0, hi);
      acc[0][j] = WMMA_BF16(a0, bb, acc[0][j]);
      acc[1][j] = WMMA_BF16(a1, bb, acc[1][j]);
    }
  }
#pragma unroll
  for (int i = 0; i < 2; ++i)
#pragma unroll
    for (int j = 0; j < 4; ++j) {
      const int n = nb + j * 16 + l15;
#pragma unroll
      for (int r = 0; r < 8; ++r) {
        int o = mg * 32 + i * 16 + r + 8 * hi;        // C/D layout: M = r + 8*hi
        __bf16 bv = f2bf(acc[i][j][r] + qkv_b[o]);
        if (o < CII)          thetaT[((size_t)b * NN + n) * CII + o] = bv;
        else if (o < 2 * CII) phiT  [((size_t)b * NN + n) * CII + (o - CII)] = bv;
        else                  g_cn  [((size_t)b * CII + (o - 2 * CII)) * NN + n] = bv;
      }
    }
}

// ---------------- K3: fused attention: S = theta*phi^T -> softmax (in LDS) -> y = P*g ----------------
// One workgroup = 16 query tokens; the 16x3136 f32 score block lives entirely in LDS
// (196KB of the 320KB WGP budget); theta block staged by the Tensor Data Mover.
__global__ __launch_bounds__(256) void k3_attention(
    const __bf16* __restrict__ thetaT, const __bf16* __restrict__ phiT,
    const __bf16* __restrict__ g_cn,   __bf16* __restrict__ yT) {
  extern __shared__ char smem[];
  __bf16* ldsA   = (__bf16*)smem;                                   // 16 x 512 bf16 (16 KB)
  float*  ldsS   = (float*)(smem + 16 * CII * sizeof(__bf16));      // 16 x 3136 f32 (196 KB)
  float*  rowMax = (float*)(smem + 16 * CII * sizeof(__bf16) + 16 * NN * sizeof(float));
  float*  rowInv = rowMax + 16;

  const int b   = blockIdx.y;
  const int q0  = blockIdx.x * 16;
  const int tid = threadIdx.x;
  const int w = tid >> 5, lane = tid & 31, hi = (lane >> 4) & 1, l15 = lane & 15;

  const __bf16* thb = thetaT + ((size_t)b * NN + q0) * CII;         // 16 contiguous rows
  const __bf16* phb = phiT + (size_t)b * NN * CII;
  const __bf16* gb  = g_cn + (size_t)b * CII * NN;

#ifdef HAVE_TDM
  // TDM: one descriptor-driven DMA stages the 16x512 bf16 theta block (16 KB,
  // contiguous) into LDS offset 0 (dynamic-LDS base; no static LDS in this kernel).
  if (w == 0) {
    const unsigned long long ga = (unsigned long long)(uintptr_t)thb;
    u32x4 g0;
    g0[0] = 1u;                                        // count=1, user descriptor
    g0[1] = 0u;                                        // lds_addr = 0
    g0[2] = (unsigned)(ga & 0xffffffffu);              // global_addr lo
    g0[3] = (unsigned)((ga >> 32) & 0x01ffffffu) | (2u << 30);  // addr hi | type=2
    i32x8 g1;
    g1[0] = (1 << 16);                                 // wg_mask=0, data_size=1 (2B)
    g1[1] = (int)(0x2000u << 16);                      // tensor_dim0 = 8192 (lo16)
    g1[2] = (int)(1u << 16);                           // tensor_dim0 hi=0 | tensor_dim1=1
    g1[3] = (int)(0x2000u << 16);                      // tile_dim0 = 8192
    g1[4] = 0;                                         // tile_dim1/2 unused
    g1[5] = 8192;                                      // tensor_dim0_stride lo32
    g1[6] = 0; g1[7] = 0;
    i32x4 zz = {0, 0, 0, 0};
#if __clang_major__ >= 23
    i32x8 z8 = {0, 0, 0, 0, 0, 0, 0, 0};
    __builtin_amdgcn_tensor_load_to_lds(g0, g1, zz, zz, z8, 0);
#else
    __builtin_amdgcn_tensor_load_to_lds(g0, g1, zz, zz, 0);
#endif
    __builtin_amdgcn_s_wait_tensorcnt(0);
  }
  __syncthreads();
#else
  { // fallback: cooperative copy, 64B per thread
    const uint4* src = (const uint4*)(thb + tid * 32);
    uint4* dst = (uint4*)(ldsA + tid * 32);
    dst[0] = src[0]; dst[1] = src[1]; dst[2] = src[2]; dst[3] = src[3];
  }
  __syncthreads();
#endif

  // ---- QK: wave w owns 64-wide key groups g = w, w+8, ...; A-frag shared by 4 WMMAs ----
  for (int g = w; g < NG; g += 8) {
    v8f acc[4];
#pragma unroll
    for (int j = 0; j < 4; ++j) acc[j] = (v8f){0.f,0.f,0.f,0.f,0.f,0.f,0.f,0.f};
    const int cb = g * 64;
    for (int k0 = 0; k0 < CII; k0 += 32) {
      v16bf a = frag_a(ldsA, CII, l15, k0, hi);
#pragma unroll
      for (int j = 0; j < 4; ++j) {
        v16bf bb = frag_b(phb, CII, cb + j * 16 + l15, k0, hi);
        acc[j] = WMMA_BF16(a, bb, acc[j]);
      }
    }
#pragma unroll
    for (int j = 0; j < 4; ++j)
#pragma unroll
      for (int r = 0; r < 8; ++r)
        ldsS[(size_t)(r + 8 * hi) * NN + cb + j * 16 + l15] = acc[j][r];
  }
  __syncthreads();

  // ---- exact softmax stats: 16 threads per query row ----
  {
    const int row = tid >> 4, lc = tid & 15;
    const float* Srow = ldsS + (size_t)row * NN;
    float mx = -3.4e38f;
    for (int m = lc; m < NN; m += 16) mx = fmaxf(mx, Srow[m]);
    for (int off = 8; off; off >>= 1) mx = fmaxf(mx, __shfl_xor(mx, off, 16));
    float s = 0.f;
    for (int m = lc; m < NN; m += 16) s += __expf(Srow[m] - mx);
    for (int off = 8; off; off >>= 1) s += __shfl_xor(s, off, 16);
    if (lc == 0) { rowMax[row] = mx; rowInv[row] = 1.f / s; }
  }
  __syncthreads();

  // ---- PV: wave w owns CI chunk [64w, 64w+64); P built on the fly from LDS scores ----
  {
    const int row = l15;
    const float mx = rowMax[row], inv = rowInv[row];
    const float* Srow = ldsS + (size_t)row * NN;
    const int cw = w * 64;
    v8f acc0 = {0.f,0.f,0.f,0.f,0.f,0.f,0.f,0.f}, acc1 = acc0, acc2 = acc0, acc3 = acc0;
    for (int k0 = 0; k0 < NN; k0 += 32) {
      v16bf a;
      const int kb1 = k0 + hi * 8, kb2 = k0 + 16 + hi * 8;
#pragma unroll
      for (int i = 0; i < 8; ++i) {
        a[i]     = f2bf(__expf(Srow[kb1 + i] - mx) * inv);
        a[i + 8] = f2bf(__expf(Srow[kb2 + i] - mx) * inv);
      }
      acc0 = WMMA_BF16(a, frag_b(gb, NN, cw + l15,      k0, hi), acc0);
      acc1 = WMMA_BF16(a, frag_b(gb, NN, cw + 16 + l15, k0, hi), acc1);
      acc2 = WMMA_BF16(a, frag_b(gb, NN, cw + 32 + l15, k0, hi), acc2);
      acc3 = WMMA_BF16(a, frag_b(gb, NN, cw + 48 + l15, k0, hi), acc3);
    }
#pragma unroll
    for (int r = 0; r < 8; ++r) {
      const size_t base = ((size_t)b * NN + q0 + r + 8 * hi) * CII;
      yT[base + cw + l15]      = f2bf(acc0[r]);
      yT[base + cw + 16 + l15] = f2bf(acc1[r]);
      yT[base + cw + 32 + l15] = f2bf(acc2[r]);
      yT[base + cw + 48 + l15] = f2bf(acc3[r]);
    }
  }
}

// ---------------- K4: z = Wz * y + bias  -> z[B,C,N] f32 ----------------
// 32(M) x 64(N) macro-tile per wave, same scheme as K2.
__global__ __launch_bounds__(256) void k4_z_gemm(
    const __bf16* __restrict__ wzb, const float* __restrict__ wz_bias,
    const __bf16* __restrict__ yT, float* __restrict__ z) {
  const int b    = blockIdx.y;
  const int w    = threadIdx.x >> 5;
  const int lane = threadIdx.x & 31;
  const int hi   = (lane >> 4) & 1;
  const int l15  = lane & 15;
  const int t  = blockIdx.x * 8 + w;                  // [0, 32*49)
  const int ng = t % NG, mg = t / NG;
  const __bf16* yb = yT + (size_t)b * NN * CII;
  const int ar0 = mg * 32 + l15;
  const int nb  = ng * 64;
  v8f acc[2][4];
#pragma unroll
  for (int i = 0; i < 2; ++i)
#pragma unroll
    for (int j = 0; j < 4; ++j) acc[i][j] = (v8f){0.f,0.f,0.f,0.f,0.f,0.f,0.f,0.f};
  for (int k0 = 0; k0 < CII; k0 += 32) {
    __builtin_prefetch(yb + (size_t)(nb + l15) * CII + k0 + 64, 0, 3);
    v16bf a0 = frag_a(wzb, CII, ar0,      k0, hi);
    v16bf a1 = frag_a(wzb, CII, ar0 + 16, k0, hi);
#pragma unroll
    for (int j = 0; j < 4; ++j) {
      v16bf bb = frag_b(yb, CII, nb + j * 16 + l15, k0, hi);
      acc[0][j] = WMMA_BF16(a0, bb, acc[0][j]);
      acc[1][j] = WMMA_BF16(a1, bb, acc[1][j]);
    }
  }
#pragma unroll
  for (int i = 0; i < 2; ++i)
#pragma unroll
    for (int j = 0; j < 4; ++j) {
      const int n = nb + j * 16 + l15;
#pragma unroll
      for (int r = 0; r < 8; ++r) {
        int c = mg * 32 + i * 16 + r + 8 * hi;
        z[((size_t)b * CC + c) * NN + n] = acc[i][j][r] + wz_bias[c];
      }
    }
}

// ---------------- K5: BN batch statistics (mean, rstd) per channel ----------------
__global__ __launch_bounds__(256) void k5_stats(
    const float* __restrict__ z, float* __restrict__ meanBuf, float* __restrict__ rstdBuf) {
  const int c = blockIdx.x;
  float s = 0.f, sq = 0.f;
  for (int i = threadIdx.x; i < BB * NN; i += 256) {
    int b = i / NN, n = i - b * NN;
    float v = z[((size_t)b * CC + c) * NN + n];
    s += v; sq += v * v;
  }
  __shared__ float sh[512];
  sh[threadIdx.x] = s; sh[256 + threadIdx.x] = sq;
  __syncthreads();
  for (int o = 128; o; o >>= 1) {
    if ((int)threadIdx.x < o) {
      sh[threadIdx.x]       += sh[threadIdx.x + o];
      sh[256 + threadIdx.x] += sh[256 + threadIdx.x + o];
    }
    __syncthreads();
  }
  if (threadIdx.x == 0) {
    const float invN = 1.f / (float)(BB * NN);
    float m   = sh[0] * invN;
    float var = sh[256] * invN - m * m;                // biased variance
    meanBuf[c] = m;
    rstdBuf[c] = rsqrtf(var + BN_EPS);
  }
}

// ---------------- K6: normalize + affine + residual ----------------
__global__ __launch_bounds__(256) void k6_norm(
    const float* __restrict__ z, const float* __restrict__ x,
    const float* __restrict__ mean, const float* __restrict__ rstd,
    const float* __restrict__ gamma, const float* __restrict__ beta,
    float* __restrict__ out) {
  const size_t i = ((size_t)blockIdx.x * 256 + threadIdx.x) * 4;
  const int c = (int)((i / NN) % CC);                  // NN % 4 == 0: 4-pack shares channel
  const float m = mean[c], sc = rstd[c] * gamma[c], bt = beta[c];
  float4 zv = *(const float4*)(z + i);
  float4 xv = *(const float4*)(x + i);
  float4 o;
  o.x = (zv.x - m) * sc + bt + xv.x;
  o.y = (zv.y - m) * sc + bt + xv.y;
  o.z = (zv.z - m) * sc + bt + xv.z;
  o.w = (zv.w - m) * sc + bt + xv.w;
  *(float4*)(out + i) = o;
}

// ---------------- host launcher ----------------
extern "C" void kernel_launch(void* const* d_in, const int* in_sizes, int n_in,
                              void* d_out, int out_size, void* d_ws, size_t ws_size,
                              hipStream_t stream) {
  const float* x       = (const float*)d_in[0];
  const float* theta_w = (const float*)d_in[1];
  const float* theta_b = (const float*)d_in[2];
  const float* phi_w   = (const float*)d_in[3];
  const float* phi_b   = (const float*)d_in[4];
  const float* g_w     = (const float*)d_in[5];
  const float* g_b     = (const float*)d_in[6];
  const float* wz_w    = (const float*)d_in[7];
  const float* wz_b    = (const float*)d_in[8];
  const float* gamma   = (const float*)d_in[9];
  const float* beta    = (const float*)d_in[10];
  float* out = (float*)d_out;

  // workspace layout (256B-aligned regions)
  char* ws = (char*)d_ws;
  const size_t OFF_WQKV  = 0;                                            // 1536x1024 bf16
  const size_t OFF_WZB   = OFF_WQKV  + (size_t)3 * CII * CC * 2;          // 1024x512 bf16
  const size_t OFF_QKVB  = OFF_WZB   + (size_t)CC * CII * 2;              // 1536 f32
  const size_t OFF_MEAN  = OFF_QKVB  + 8192;                              // 1024 f32
  const size_t OFF_RSTD  = OFF_MEAN  + (size_t)CC * 4;                    // 1024 f32
  const size_t OFF_XT    = OFF_RSTD  + (size_t)CC * 4 + 256;              // B*N*C bf16
  const size_t OFF_THETA = OFF_XT    + (size_t)BB * NN * CC * 2;          // B*N*CI bf16
  const size_t OFF_PHI   = OFF_THETA + (size_t)BB * NN * CII * 2;
  const size_t OFF_G     = OFF_PHI   + (size_t)BB * NN * CII * 2;
  const size_t OFF_Y     = OFF_G     + (size_t)BB * NN * CII * 2;
  const size_t OFF_Z     = OFF_Y     + (size_t)BB * NN * CII * 2;         // B*C*N f32

  __bf16* wqkv   = (__bf16*)(ws + OFF_WQKV);
  __bf16* wzb    = (__bf16*)(ws + OFF_WZB);
  float*  qkv_b  = (float*) (ws + OFF_QKVB);
  float*  meanB  = (float*) (ws + OFF_MEAN);
  float*  rstdB  = (float*) (ws + OFF_RSTD);
  __bf16* xT     = (__bf16*)(ws + OFF_XT);
  __bf16* thetaT = (__bf16*)(ws + OFF_THETA);
  __bf16* phiT   = (__bf16*)(ws + OFF_PHI);
  __bf16* g_cn   = (__bf16*)(ws + OFF_G);
  __bf16* yT     = (__bf16*)(ws + OFF_Y);
  float*  z      = (float*) (ws + OFF_Z);

  // K0: pack weights (1536*1024 + 1024*512 elements)
  k0_prep<<<(3 * CII * CC + CC * CII) / 256, 256, 0, stream>>>(
      theta_w, phi_w, g_w, theta_b, phi_b, g_b, wz_w, wqkv, wzb, qkv_b);

  // K1: transpose-cast x -> xT
  k1_transpose_cast<<<dim3(CC / 32, NN / 32, BB), 256, 0, stream>>>(x, xT);

  // K2: QKV projection GEMM (48x49 wave-tiles / batch, 8 waves per WG)
  k2_qkv_gemm<<<dim3((48 * NG) / 8, BB), 256, 0, stream>>>(
      wqkv, qkv_b, xT, thetaT, phiT, g_cn);

  // K3: fused attention; 217216 B dynamic LDS (16KB A + 196KB scores + stats)
  const size_t smemK3 = (size_t)16 * CII * 2 + (size_t)16 * NN * 4 + 128;
  k3_attention<<<dim3(NT, BB), 256, smemK3, stream>>>(thetaT, phiT, g_cn, yT);

  // K4: output projection GEMM (32x49 wave-tiles / batch)
  k4_z_gemm<<<dim3((32 * NG) / 8, BB), 256, 0, stream>>>(wzb, wz_b, yT, z);

  // K5: BN stats per channel
  k5_stats<<<CC, 256, 0, stream>>>(z, meanB, rstdB);

  // K6: normalize + residual (4 elems/thread)
  k6_norm<<<(size_t)BB * CC * NN / 4 / 256, 256, 0, stream>>>(
      z, x, meanB, rstdB, gamma, beta, out);
}
